// ClusterBasedGraphCompression_35021163331786
// MI455X (gfx1250) — matrix operand order repaired and checked by reading, not compiled, not run on hardware
//
#include <hip/hip_runtime.h>

// ---------------------------------------------------------------------------
// ClusterBasedGraphCompression for MI455X (gfx1250, wave32, WMMA)
//
// Shapes: B=8, N=8192, IN=1024, D=512, C=64
// Pipeline:
//   P  : prep  -> W_nodeT bf16 [512][1024], W_clusterT bf16 [512][512], cc bf16
//   A  : ccp   = cc @ Wc + bc            (WMMA bf16, 64x512)
//   B  : proj  = X @ Wn + bn             (WMMA bf16, LDS-staged A, 128x256 tile)
//   C  : sims  = proj . ccp              (WMMA bf16, stored [b][c][n] in d_out)
//   D  : softmax over n (in place in d_out) + bf16 copy of weights in ws
//   E  : agg   = weights . proj + ccp    (WMMA bf16, LDS-transposed B operand)
//   F  : LayerNorm over d -> d_out[0:262144]
// ---------------------------------------------------------------------------

typedef __attribute__((ext_vector_type(16))) __bf16 v16bf;
typedef __attribute__((ext_vector_type(8)))  __bf16 v8bf;
typedef __attribute__((ext_vector_type(4)))  __bf16 v4bf;
typedef __attribute__((ext_vector_type(8)))  float  v8f;
typedef __attribute__((ext_vector_type(4)))  float  v4f;

#define BATCH   8
#define NNODE   8192
#define INDIM   1024
#define ODIM    512
#define NCLUS   64

static __device__ __forceinline__ v8f wmma_bf16(v16bf a, v16bf b, v8f c) {
  // D = A(16x32) * B(32x16) + C, fp32 accumulate
  return __builtin_amdgcn_wmma_f32_16x16x32_bf16(
      /*neg_a=*/false, a, /*neg_b=*/false, b,
      /*c_mod=*/(short)0, c, /*reuse_a=*/false, /*reuse_b=*/false);
}

static __device__ __forceinline__ v16bf combine16(v8bf lo, v8bf hi) {
  v16bf r;
#pragma unroll
  for (int i = 0; i < 8; ++i) { r[i] = lo[i]; r[8 + i] = hi[i]; }
  return r;
}

// load 8 consecutive fp32 and round to bf16
static __device__ __forceinline__ v8bf cvt8(const float* __restrict__ p) {
  v4f a = *(const v4f*)p;
  v4f b = *(const v4f*)(p + 4);
  v8bf r;
#pragma unroll
  for (int i = 0; i < 4; ++i) { r[i] = (__bf16)a[i]; r[4 + i] = (__bf16)b[i]; }
  return r;
}

static __device__ __forceinline__ v16bf load_bfrag(const __bf16* __restrict__ p) {
  // 16 contiguous bf16 (32B), 16B-aligned
  return combine16(*(const v8bf*)p, *(const v8bf*)(p + 8));
}

// ---------------------------------------------------------------------------
// P: transpose + bf16-convert weight matrices; convert cluster centers
// ---------------------------------------------------------------------------
__global__ void k_prep(const float* __restrict__ Wn, const float* __restrict__ Wc,
                       const float* __restrict__ CC,
                       __bf16* __restrict__ WnT, __bf16* __restrict__ WcT,
                       __bf16* __restrict__ CCb) {
  const long T1 = (long)ODIM * INDIM;      // W_nodeT
  const long T2 = (long)ODIM * ODIM;       // W_clusterT
  const long T3 = (long)NCLUS * ODIM;      // cc bf16
  long i = (long)blockIdx.x * blockDim.x + threadIdx.x;
  if (i < T1) {
    long dout = i / INDIM, din = i % INDIM;
    WnT[i] = (__bf16)Wn[din * ODIM + dout];
  } else if (i < T1 + T2) {
    long j = i - T1;
    long dout = j / ODIM, din = j % ODIM;
    WcT[j] = (__bf16)Wc[din * ODIM + dout];
  } else if (i < T1 + T2 + T3) {
    long j = i - T1 - T2;
    CCb[j] = (__bf16)CC[j];
  }
}

// ---------------------------------------------------------------------------
// A: ccp = cc @ Wc + bc   -> fp32 + bf16 copies.  grid=32 blocks x 128 thr.
// ---------------------------------------------------------------------------
__global__ void k_ccp(const __bf16* __restrict__ CCb, const __bf16* __restrict__ WcT,
                      const float* __restrict__ bc,
                      float* __restrict__ CCPF, __bf16* __restrict__ CCPB) {
  const int wave = threadIdx.x >> 5, lane = threadIdx.x & 31;
  const int g = lane >> 4, nl = lane & 15;
  const int c0 = wave * 16;
  const int d0 = blockIdx.x * 16;
  v8f acc;
  const float bv = bc[d0 + nl];
#pragma unroll
  for (int r = 0; r < 8; ++r) acc[r] = bv;
  const __bf16* arow = CCb + (size_t)(c0 + nl) * ODIM;
  for (int k0 = 0; k0 < ODIM; k0 += 32) {
    v16bf a = combine16(*(const v8bf*)(arow + k0 + g * 16),
                        *(const v8bf*)(arow + k0 + g * 16 + 8));
    v16bf b = load_bfrag(WcT + (size_t)(d0 + nl) * ODIM + k0 + g * 16);
    acc = wmma_bf16(a, b, acc);
  }
#pragma unroll
  for (int r = 0; r < 8; ++r) {
    int c = c0 + g * 8 + r, d = d0 + nl;
    float v = acc[r];
    CCPF[c * ODIM + d] = v;
    CCPB[c * ODIM + d] = (__bf16)v;
  }
}

// ---------------------------------------------------------------------------
// B: node_proj = X @ Wn + bn (bf16 out).  grid = (2 colgrp, 512 rowgrp) x 256.
// Block tile 128 rows x 256 cols; per K-step the 128x32 fp32 X slab is staged
// once into LDS as bf16 so X is only re-read 2x from L2 (HBM sees it once).
// Each wave: 16 rows x 256 cols = 16 WMMA accumulators.
// Epilogue repacks through LDS in 4 rounds for 16B-coalesced bf16 stores.
// ---------------------------------------------------------------------------
__global__ void k_node_proj(const float* __restrict__ X, const float* __restrict__ bn,
                            const __bf16* __restrict__ WnT, __bf16* __restrict__ P) {
  __shared__ __bf16 lds[128 * 80];  // staging uses [128][40], repack uses [128][80]
  const int tid = threadIdx.x;
  const int wave = tid >> 5, lane = tid & 31;
  const int g = lane >> 4, nl = lane & 15;
  const int nd0 = blockIdx.x * 256;
  const int m0 = blockIdx.y * 128;

  v8f acc[16];
#pragma unroll
  for (int t = 0; t < 16; ++t) {
    float bv = bn[nd0 + t * 16 + nl];
#pragma unroll
    for (int r = 0; r < 8; ++r) acc[t][r] = bv;
  }

  const int srow = tid >> 1, shalf = tid & 1;           // staging: 16 fp32 per thread
  const float* xsrc = X + (size_t)(m0 + srow) * INDIM + shalf * 16;
  __bf16* sdst = &lds[srow * 40 + shalf * 16];
  const __bf16* afrag0 = &lds[(wave * 16 + nl) * 40 + g * 16];      // k: g*16..g*16+7
  const __bf16* afrag1 = &lds[(wave * 16 + nl) * 40 + g * 16 + 8]; // k: g*16+8..g*16+15

  for (int k0 = 0; k0 < INDIM; k0 += 32) {
    __syncthreads();  // protect previous iteration's LDS reads
    __builtin_prefetch(xsrc + k0 + 64, 0, 0);  // global_prefetch_b8 (next slabs)
    *(v8bf*)sdst = cvt8(xsrc + k0);
    *(v8bf*)(sdst + 8) = cvt8(xsrc + k0 + 8);
    __syncthreads();
    v16bf a = combine16(*(const v8bf*)afrag0, *(const v8bf*)afrag1);
#pragma unroll
    for (int t = 0; t < 16; ++t) {
      v16bf b = load_bfrag(WnT + (size_t)(nd0 + t * 16 + nl) * INDIM + k0 + g * 16);
      acc[t] = wmma_bf16(a, b, acc[t]);
    }
  }

  // repack 128x256 bf16 through LDS in 4 rounds of 64 cols
  const int orow = tid >> 1, ohalf = tid & 1;
#pragma unroll
  for (int rr = 0; rr < 4; ++rr) {
    __syncthreads();
#pragma unroll
    for (int tt = 0; tt < 4; ++tt) {
      const int t = rr * 4 + tt;
      const int col = tt * 16 + nl;
#pragma unroll
      for (int r = 0; r < 8; ++r) {
        int mr = wave * 16 + g * 8 + r;
        lds[mr * 80 + col] = (__bf16)acc[t][r];
      }
    }
    __syncthreads();
    const __bf16* src = &lds[orow * 80 + ohalf * 32];
    __bf16* dst = P + (size_t)(m0 + orow) * ODIM + nd0 + rr * 64 + ohalf * 32;
#pragma unroll
    for (int q = 0; q < 4; ++q)
      *(v8bf*)(dst + q * 8) = *(const v8bf*)(src + q * 8);
  }
}

// ---------------------------------------------------------------------------
// C: sims[b][c][n] = sum_d proj[b][n][d]*ccp[c][d]
// Computed in [n][c] orientation: A = proj rows (M=n), B = ccp rows (N'=c).
// D-tile store lands contiguous in n.  grid = (64 rowgrp, 8 batch) x 256.
// ---------------------------------------------------------------------------
__global__ void k_sims(const __bf16* __restrict__ P, const __bf16* __restrict__ CCPB,
                       float* __restrict__ S) {
  const int wave = threadIdx.x >> 5, lane = threadIdx.x & 31;
  const int g = lane >> 4, nl = lane & 15;
  const int b = blockIdx.y;
  const int n0 = blockIdx.x * 128 + wave * 16;

  v8f acc[4];
#pragma unroll
  for (int t = 0; t < 4; ++t)
#pragma unroll
    for (int r = 0; r < 8; ++r) acc[t][r] = 0.0f;

  const __bf16* arow = P + ((size_t)b * NNODE + n0 + nl) * ODIM;
  for (int k0 = 0; k0 < ODIM; k0 += 32) {
    v16bf a = combine16(*(const v8bf*)(arow + k0 + g * 16),
                        *(const v8bf*)(arow + k0 + g * 16 + 8));
#pragma unroll
    for (int t = 0; t < 4; ++t) {
      v16bf bb = load_bfrag(CCPB + (size_t)(t * 16 + nl) * ODIM + k0 + g * 16);
      acc[t] = wmma_bf16(a, bb, acc[t]);
    }
  }
#pragma unroll
  for (int t = 0; t < 4; ++t) {
    int c = t * 16 + nl;
    float* dst = S + ((size_t)b * NCLUS + c) * NNODE + n0 + g * 8;
    v4f lo, hi;
#pragma unroll
    for (int r = 0; r < 4; ++r) { lo[r] = acc[t][r]; hi[r] = acc[t][4 + r]; }
    *(v4f*)dst = lo;
    *(v4f*)(dst + 4) = hi;
  }
}

// ---------------------------------------------------------------------------
// D: row softmax over n (8192) per (b,c); in-place fp32 in d_out + bf16 copy.
// ---------------------------------------------------------------------------
__global__ void k_softmax(float* __restrict__ S, const unsigned char* __restrict__ mask,
                          __bf16* __restrict__ WB) {
  __shared__ float red[256];
  const int bc = blockIdx.x;
  const int b = bc >> 6;
  float* row = S + (size_t)bc * NNODE;
  const unsigned char* mrow = mask + (size_t)b * NNODE;
  const int tid = threadIdx.x;
  const float NEG_INF = -__builtin_inff();

  float mx = NEG_INF;
  for (int j = tid; j < NNODE; j += 256) {
    float v = mrow[j] ? NEG_INF : row[j];
    mx = fmaxf(mx, v);
  }
  red[tid] = mx;
  __syncthreads();
  for (int s = 128; s > 0; s >>= 1) {
    if (tid < s) red[tid] = fmaxf(red[tid], red[tid + s]);
    __syncthreads();
  }
  mx = red[0];
  __syncthreads();

  float sum = 0.0f;
  for (int j = tid; j < NNODE; j += 256) {
    float v = mrow[j] ? NEG_INF : row[j];
    sum += expf(v - mx);
  }
  red[tid] = sum;
  __syncthreads();
  for (int s = 128; s > 0; s >>= 1) {
    if (tid < s) red[tid] += red[tid + s];
    __syncthreads();
  }
  float inv = 1.0f / red[0];

  for (int j = tid; j < NNODE; j += 256) {
    float v = mrow[j] ? NEG_INF : row[j];
    float w = expf(v - mx) * inv;
    row[j] = w;
    WB[(size_t)bc * NNODE + j] = (__bf16)w;
  }
}

// ---------------------------------------------------------------------------
// E: compressed_pre[b][c][d] = sum_n w[b][c][n]*proj[b][n][d] + ccp[c][d]
// A = weights rows (K=n contiguous). B operand needs K=n contiguous per d-lane:
// transpose 32x16 proj tiles through LDS each K-step.
// grid = (32 colgrp, 8 batch) x 128 threads (4 waves = 4 m-tiles over C=64).
// ---------------------------------------------------------------------------
__global__ void k_aggregate(const __bf16* __restrict__ WB, const __bf16* __restrict__ P,
                            const float* __restrict__ CCPF, float* __restrict__ OUTPRE) {
  __shared__ __bf16 ldsB[16 * 40];  // [d=16][n=32], pad to 40 (80B stride)
  const int tid = threadIdx.x;
  const int wave = tid >> 5, lane = tid & 31;
  const int g = lane >> 4, nl = lane & 15;
  const int b = blockIdx.y;
  const int d0 = blockIdx.x * 16;
  const int rown = tid >> 2, part = tid & 3;

  v8f acc;
#pragma unroll
  for (int r = 0; r < 8; ++r) acc[r] = 0.0f;

  const __bf16* arow = WB + ((size_t)b * NCLUS + wave * 16 + nl) * NNODE;

  for (int k0 = 0; k0 < NNODE; k0 += 32) {
    __syncthreads();  // protect previous iteration's reads
    {
      const __bf16* src = P + ((size_t)b * NNODE + k0 + rown) * ODIM + d0 + part * 4;
      v4bf v = *(const v4bf*)src;
#pragma unroll
      for (int j = 0; j < 4; ++j) ldsB[(part * 4 + j) * 40 + rown] = v[j];
    }
    __syncthreads();
    v16bf a = combine16(*(const v8bf*)(arow + k0 + g * 16),
                        *(const v8bf*)(arow + k0 + g * 16 + 8));
    const __bf16* bp = &ldsB[nl * 40 + g * 16];
    v16bf bb = combine16(*(const v8bf*)bp, *(const v8bf*)(bp + 8));
    acc = wmma_bf16(a, bb, acc);
  }

#pragma unroll
  for (int r = 0; r < 8; ++r) {
    int c = wave * 16 + g * 8 + r;
    int d = d0 + nl;
    OUTPRE[((size_t)b * NCLUS + c) * ODIM + d] = acc[r] + CCPF[c * ODIM + d];
  }
}

// ---------------------------------------------------------------------------
// F: LayerNorm over d=512 per (b,c) row -> final output
// ---------------------------------------------------------------------------
__global__ void k_layernorm(const float* __restrict__ PRE, const float* __restrict__ gamma,
                            const float* __restrict__ beta, float* __restrict__ OUT) {
  __shared__ float s1[256], s2[256];
  const int bc = blockIdx.x, tid = threadIdx.x;
  const float* row = PRE + (size_t)bc * ODIM;
  float x0 = row[tid], x1 = row[tid + 256];
  s1[tid] = x0 + x1;
  s2[tid] = x0 * x0 + x1 * x1;
  __syncthreads();
  for (int s = 128; s > 0; s >>= 1) {
    if (tid < s) { s1[tid] += s1[tid + s]; s2[tid] += s2[tid + s]; }
    __syncthreads();
  }
  const float mu = s1[0] * (1.0f / ODIM);
  const float var = s2[0] * (1.0f / ODIM) - mu * mu;
  const float rs = rsqrtf(var + 1e-5f);
  float* orow = OUT + (size_t)bc * ODIM;
  orow[tid] = (x0 - mu) * rs * gamma[tid] + beta[tid];
  orow[tid + 256] = (x1 - mu) * rs * gamma[tid + 256] + beta[tid + 256];
}

// ---------------------------------------------------------------------------
extern "C" void kernel_launch(void* const* d_in, const int* in_sizes, int n_in,
                              void* d_out, int out_size, void* d_ws, size_t ws_size,
                              hipStream_t stream) {
  const float* X            = (const float*)d_in[0];  // [8,8192,1024]
  const unsigned char* mask = (const unsigned char*)d_in[1];  // [8,8192] bool
  const float* CC           = (const float*)d_in[2];  // [64,512]
  const float* Wn           = (const float*)d_in[3];  // [1024,512]
  const float* bn           = (const float*)d_in[4];  // [512]
  const float* Wc           = (const float*)d_in[5];  // [512,512]
  const float* bcl          = (const float*)d_in[6];  // [512]
  const float* gam          = (const float*)d_in[7];  // [512]
  const float* bet          = (const float*)d_in[8];  // [512]

  float* out = (float*)d_out;                 // [262144 compressed][4194304 weights]
  float* simsraw = out + (size_t)BATCH * NCLUS * ODIM;  // weights region as scratch

  // workspace carve (256B aligned chunks)
  char* ws = (char*)d_ws;
  size_t off = 0;
  auto carve = [&](size_t bytes) {
    char* p = ws + off;
    off += (bytes + 255) & ~(size_t)255;
    return p;
  };
  __bf16* Pbf  = (__bf16*)carve((size_t)BATCH * NNODE * ODIM * 2);  // 64 MB
  __bf16* WBbf = (__bf16*)carve((size_t)BATCH * NCLUS * NNODE * 2); // 8 MB
  float*  CCPF = (float*)carve((size_t)NCLUS * ODIM * 4);
  __bf16* CCPB = (__bf16*)carve((size_t)NCLUS * ODIM * 2);
  __bf16* WnT  = (__bf16*)carve((size_t)ODIM * INDIM * 2);
  __bf16* WcT  = (__bf16*)carve((size_t)ODIM * ODIM * 2);
  __bf16* CCb  = (__bf16*)carve((size_t)NCLUS * ODIM * 2);
  float*  CPRE = (float*)carve((size_t)BATCH * NCLUS * ODIM * 4);
  (void)ws_size; (void)in_sizes; (void)n_in; (void)out_size;

  const long prepN = (long)ODIM * INDIM + (long)ODIM * ODIM + (long)NCLUS * ODIM;
  k_prep<<<(int)((prepN + 255) / 256), 256, 0, stream>>>(Wn, Wc, CC, WnT, WcT, CCb);
  k_ccp<<<ODIM / 16, 128, 0, stream>>>(CCb, WcT, bcl, CCPF, CCPB);
  k_node_proj<<<dim3(ODIM / 256, (BATCH * NNODE) / 128), 256, 0, stream>>>(X, bn, WnT, Pbf);
  k_sims<<<dim3(NNODE / 128, BATCH), 256, 0, stream>>>(Pbf, CCPB, simsraw);
  k_softmax<<<BATCH * NCLUS, 256, 0, stream>>>(simsraw, mask, WBbf);
  k_aggregate<<<dim3(ODIM / 16, BATCH), 128, 0, stream>>>(WBbf, Pbf, CCPF, CPRE);
  k_layernorm<<<BATCH * NCLUS, 256, 0, stream>>>(CPRE, gam, bet, out);
}
